// MSAPairWeightedAveraging_82420422410871
// MI455X (gfx1250) — matrix operand order repaired and checked by reading, not compiled, not run on hardware
//
#include <hip/hip_runtime.h>
#include <hip/hip_bf16.h>

// MSAPairWeightedAveraging for MI455X (gfx1250): bf16 WMMA everywhere.
// All GEMM operands are fed via contiguous 16B/32B vector loads (A row-major,
// B pre-packed into WMMA fragment layout), 32-row macro-tiles per wave.
typedef __bf16 bf16;
typedef __attribute__((ext_vector_type(16))) __bf16 v16bf;
typedef __attribute__((ext_vector_type(8)))  __bf16 v8bf;
typedef __attribute__((ext_vector_type(8)))  float  v8f;

#define SDIM 256
#define NDIM 512
#define CMD  256
#define CZD  128
#define HD   8
#define CD   32
#define HC   256

// ---------------- wave32 reductions ----------------
__device__ __forceinline__ float wave_sum(float v) {
#pragma unroll
  for (int m = 16; m >= 1; m >>= 1) v += __shfl_xor(v, m, 32);
  return v;
}
__device__ __forceinline__ float wave_max(float v) {
#pragma unroll
  for (int m = 16; m >= 1; m >>= 1) v = fmaxf(v, __shfl_xor(v, m, 32));
  return v;
}

// ---------------- WMMA helpers (contiguous k order: k = 16*(lane/16) + s) ----------------
// A 16x32 bf16, row-major source with leading dim ld.
__device__ __forceinline__ v16bf load_frag_a(const bf16* __restrict__ src, int ld,
                                             int row0, int k0, int lane) {
  int r  = row0 + (lane & 15);
  int kb = k0 + ((lane >> 4) ? 16 : 0);
  const bf16* p = src + (size_t)r * ld + kb;
  v8bf lo = *(const v8bf*)(p);        // k = kb .. kb+7
  v8bf hi = *(const v8bf*)(p + 8);    // k = kb+8 .. kb+15
  v16bf out;
#pragma unroll
  for (int i = 0; i < 8; ++i) { out[i] = lo[i]; out[i + 8] = hi[i]; }
  return out;
}

// Packed B fragment: P[frag][lane][16] contiguous (32B per lane).
__device__ __forceinline__ v16bf load_frag_bp(const bf16* __restrict__ P,
                                              size_t frag, int lane) {
  return *(const v16bf*)(P + (frag * 32 + lane) * 16);
}

__device__ __forceinline__ v8f wmma_bf16(v16bf a, v16bf b, v8f c) {
  return __builtin_amdgcn_wmma_f32_16x16x32_bf16(false, a, false, b, (short)0, c,
                                                 false, false);
}

// ---------------- kernel 1: msa LayerNorm -> bf16 ----------------
__global__ void k_msa_ln(const float* __restrict__ x, const float* __restrict__ g,
                         const float* __restrict__ b, bf16* __restrict__ out) {
  int wave = (blockIdx.x * blockDim.x + threadIdx.x) >> 5;
  int lane = threadIdx.x & 31;
  const float* row = x + (size_t)wave * CMD;
  float4 p0 = *(const float4*)(row + lane * 8);
  float4 p1 = *(const float4*)(row + lane * 8 + 4);
  float v[8] = {p0.x, p0.y, p0.z, p0.w, p1.x, p1.y, p1.z, p1.w};
  float s = 0.f, ss = 0.f;
#pragma unroll
  for (int i = 0; i < 8; ++i) { s += v[i]; ss += v[i] * v[i]; }
  s = wave_sum(s); ss = wave_sum(ss);
  float mu  = s * (1.0f / CMD);
  float inv = rsqrtf(ss * (1.0f / CMD) - mu * mu + 1e-5f);
  v8bf ov;
#pragma unroll
  for (int i = 0; i < 8; ++i) {
    int c = lane * 8 + i;
    ov[i] = (bf16)(((v[i] - mu) * inv) * g[c] + b[c]);
  }
  *(v8bf*)(out + (size_t)wave * CMD + lane * 8) = ov;
}

// ---------------- kernel 2: pack f32 weight [K][Nn] -> WMMA B fragments -----
// frag index = kt * (Nn/16) + nt ; lanes 0-15: n=nt*16+lane, k=kt*32+i
// lanes 16-31: n = nt*16+lane-16, k = kt*32+16+i
__global__ void k_pack_w(const float* __restrict__ W, bf16* __restrict__ Wp,
                         int K, int Nn) {
  int nt_count = Nn / 16;
  int tile = (blockIdx.x * blockDim.x + threadIdx.x) >> 5;
  int lane = threadIdx.x & 31;
  if (tile >= (K / 32) * nt_count) return;
  int kt = tile / nt_count, nt = tile % nt_count;
  int n  = nt * 16 + (lane & 15);
  int kb = kt * 32 + ((lane >> 4) ? 16 : 0);
  v16bf frag;
#pragma unroll
  for (int i = 0; i < 16; ++i) frag[i] = (bf16)W[(size_t)(kb + i) * Nn + n];
  *(v16bf*)(Wp + ((size_t)tile * 32 + lane) * 16) = frag;
}

// ---------------- kernel 3: fused v/g GEMM, 32-row macro-tile ---------------
// v is stored into WMMA-B-packed layout for k_avg_gate:
//   frag(s,h,jt,ch) = ((s*HD + h)*16 + jt)*2 + ch
// g is stored row-major bf16 [S*N][HC].
__global__ void k_gemm_vg(const bf16* __restrict__ m, const bf16* __restrict__ WvP,
                          const bf16* __restrict__ WgP, const float* __restrict__ bv,
                          const float* __restrict__ bg, bf16* __restrict__ vP,
                          bf16* __restrict__ g_out) {
  int wave = (blockIdx.x * blockDim.x + threadIdx.x) >> 5;
  int lane = threadIdx.x & 31;
  const int col_tiles = HC / 16;                 // 16
  int rt = wave / col_tiles, ct = wave % col_tiles;
  int row0 = rt * 32;                            // 32 rows per wave
  v8f av0 = {}, av1 = {}, ag0 = {}, ag1 = {};
#pragma unroll
  for (int kt = 0; kt < CMD / 32; ++kt) {
    __builtin_prefetch(m + (size_t)(row0 + (lane & 15)) * CMD + ((kt + 1) & 7) * 32, 0, 1);
    v16bf a0 = load_frag_a(m, CMD, row0,      kt * 32, lane);
    v16bf a1 = load_frag_a(m, CMD, row0 + 16, kt * 32, lane);
    v16bf b0 = load_frag_bp(WvP, (size_t)kt * col_tiles + ct, lane);
    v16bf b1 = load_frag_bp(WgP, (size_t)kt * col_tiles + ct, lane);
    av0 = wmma_bf16(a0, b0, av0);
    av1 = wmma_bf16(a1, b0, av1);
    ag0 = wmma_bf16(a0, b1, ag0);
    ag1 = wmma_bf16(a1, b1, ag1);
  }
  int col   = ct * 16 + (lane & 15);
  float biv = bv[col], big = bg[col];
  int h = ct >> 1, ch = ct & 1;
#pragma unroll
  for (int t = 0; t < 2; ++t) {
    v8f accv = t ? av1 : av0;
    v8f accg = t ? ag1 : ag0;
    int r0   = row0 + t * 16;
    int rb   = r0 + ((lane >> 4) ? 8 : 0);
    // ---- v: one contiguous 16B store into packed-B layout ----
    int s  = r0 >> 9;                            // row / 512
    int j0 = (r0 & 511) + ((lane >> 4) ? 8 : 0);
    size_t frag = ((size_t)(s * HD + h) * 16 + (j0 >> 5)) * 2 + ch;
    int slot = (lane & 15) + (((j0 >> 4) & 1) << 4);
    v8bf pv;
#pragma unroll
    for (int r = 0; r < 8; ++r) pv[r] = (bf16)(accv[r] + biv);
    *(v8bf*)(vP + (frag * 32 + slot) * 16 + (j0 & 15)) = pv;
    // ---- g: sigmoid, row-major bf16 ----
#pragma unroll
    for (int r = 0; r < 8; ++r) {
      float gg = accg[r] + big;
      g_out[(size_t)(rb + r) * HC + col] = (bf16)(1.0f / (1.0f + __expf(-gg)));
    }
  }
}

// ---------------- kernel 4: pair LayerNorm + Wb projection ------------------
__global__ void k_pair_ln_b(const float* __restrict__ z, const float* __restrict__ gz,
                            const float* __restrict__ bz, const float* __restrict__ Wb,
                            const float* __restrict__ bb, float* __restrict__ bout) {
  int wave = (blockIdx.x * blockDim.x + threadIdx.x) >> 5;
  int lane = threadIdx.x & 31;
  const float* row = z + (size_t)wave * CZD;
  float4 p = *(const float4*)(row + lane * 4);
  float v[4] = {p.x, p.y, p.z, p.w};
  float s = 0.f, ss = 0.f;
#pragma unroll
  for (int i = 0; i < 4; ++i) { s += v[i]; ss += v[i] * v[i]; }
  s = wave_sum(s); ss = wave_sum(ss);
  float mu  = s * (1.0f / CZD);
  float inv = rsqrtf(ss * (1.0f / CZD) - mu * mu + 1e-5f);
  float nrm[4];
#pragma unroll
  for (int i = 0; i < 4; ++i) {
    int c = lane * 4 + i;
    nrm[i] = (v[i] - mu) * inv * gz[c] + bz[c];
  }
  float hacc[HD];
#pragma unroll
  for (int h = 0; h < HD; ++h) {
    float part = 0.f;
#pragma unroll
    for (int i = 0; i < 4; ++i) part += nrm[i] * Wb[(size_t)(lane * 4 + i) * HD + h];
    hacc[h] = wave_sum(part);
  }
  if (lane < HD) bout[(size_t)wave * HD + lane] = hacc[lane] + bb[lane];
}

// ---------------- kernel 5: softmax over j, store bf16 in [h][i][j] ---------
__global__ void k_softmax(const float* __restrict__ bin, bf16* __restrict__ w_out) {
  int wave = (blockIdx.x * blockDim.x + threadIdx.x) >> 5;
  int lane = threadIdx.x & 31;
  int i = wave >> 3;
  int h = wave & 7;
  float vals[16];
  float mx = -1e30f;
#pragma unroll
  for (int t = 0; t < 16; ++t) {
    int j = lane + t * 32;
    vals[t] = bin[((size_t)i * NDIM + j) * HD + h];
    mx = fmaxf(mx, vals[t]);
  }
  mx = wave_max(mx);
  float sum = 0.f;
#pragma unroll
  for (int t = 0; t < 16; ++t) { vals[t] = __expf(vals[t] - mx); sum += vals[t]; }
  sum = wave_sum(sum);
  float rs = 1.0f / sum;
#pragma unroll
  for (int t = 0; t < 16; ++t) {
    int j = lane + t * 32;
    w_out[(size_t)h * NDIM * NDIM + (size_t)i * NDIM + j] = (bf16)(vals[t] * rs);
  }
}

// ---------------- kernel 6: o = g * (w @ v) per (s,h), 32x32 macro-tile -----
__global__ void k_avg_gate(const bf16* __restrict__ w, const bf16* __restrict__ vP,
                           const bf16* __restrict__ g, bf16* __restrict__ o) {
  int wave = (blockIdx.x * blockDim.x + threadIdx.x) >> 5;
  int lane = threadIdx.x & 31;
  int it = wave & 15;            // N/32 = 16 i-macrotiles
  int h  = (wave >> 4) & 7;
  int s  = wave >> 7;
  int i0 = it * 32;
  const bf16* wA = w + (size_t)h * NDIM * NDIM;   // [i][j], ld=NDIM
  size_t fbase = (size_t)(s * HD + h) * 16 * 2;   // frag(s,h,jt,ch)
  v8f a00 = {}, a01 = {}, a10 = {}, a11 = {};
  for (int jt = 0; jt < NDIM / 32; ++jt) {
    __builtin_prefetch(wA + (size_t)(i0 + (lane & 15)) * NDIM + ((jt + 1) & 15) * 32, 0, 1);
    v16bf fa0 = load_frag_a(wA, NDIM, i0,      jt * 32, lane);
    v16bf fa1 = load_frag_a(wA, NDIM, i0 + 16, jt * 32, lane);
    v16bf fb0 = load_frag_bp(vP, fbase + jt * 2 + 0, lane);
    v16bf fb1 = load_frag_bp(vP, fbase + jt * 2 + 1, lane);
    a00 = wmma_bf16(fa0, fb0, a00);
    a01 = wmma_bf16(fa0, fb1, a01);
    a10 = wmma_bf16(fa1, fb0, a10);
    a11 = wmma_bf16(fa1, fb1, a11);
  }
  int cl = lane & 15;
#pragma unroll
  for (int t = 0; t < 2; ++t) {
    v8f ac0 = t ? a10 : a00;
    v8f ac1 = t ? a11 : a01;
    int rbase = i0 + t * 16 + ((lane >> 4) ? 8 : 0);
#pragma unroll
    for (int r = 0; r < 8; ++r) {
      size_t base = ((size_t)s * NDIM + (rbase + r)) * HC + h * CD;
      float g0 = (float)g[base + cl];
      float g1 = (float)g[base + 16 + cl];
      o[base + cl]      = (bf16)(ac0[r] * g0);
      o[base + 16 + cl] = (bf16)(ac1[r] * g1);
    }
  }
}

// ---------------- kernel 7: final projection, 32-row macro-tile, f32 out ----
__global__ void k_gemm_out(const bf16* __restrict__ o, const bf16* __restrict__ WpP,
                           float* __restrict__ out) {
  int wave = (blockIdx.x * blockDim.x + threadIdx.x) >> 5;
  int lane = threadIdx.x & 31;
  const int col_tiles = CMD / 16;
  int rt = wave / col_tiles, ct = wave % col_tiles;
  int row0 = rt * 32;
  v8f acc0 = {}, acc1 = {};
#pragma unroll
  for (int kt = 0; kt < HC / 32; ++kt) {
    __builtin_prefetch(o + (size_t)(row0 + (lane & 15)) * HC + ((kt + 1) & 7) * 32, 0, 1);
    v16bf a0 = load_frag_a(o, HC, row0,      kt * 32, lane);
    v16bf a1 = load_frag_a(o, HC, row0 + 16, kt * 32, lane);
    v16bf b  = load_frag_bp(WpP, (size_t)kt * col_tiles + ct, lane);
    acc0 = wmma_bf16(a0, b, acc0);
    acc1 = wmma_bf16(a1, b, acc1);
  }
  int col = ct * 16 + (lane & 15);
#pragma unroll
  for (int t = 0; t < 2; ++t) {
    v8f acc = t ? acc1 : acc0;
    int rbase = row0 + t * 16 + ((lane >> 4) ? 8 : 0);
#pragma unroll
    for (int r = 0; r < 8; ++r)
      out[(size_t)(rbase + r) * CMD + col] = acc[r];
  }
}

// ---------------- host launcher ----------------
extern "C" void kernel_launch(void* const* d_in, const int* in_sizes, int n_in,
                              void* d_out, int out_size, void* d_ws, size_t ws_size,
                              hipStream_t stream) {
  const float* msa   = (const float*)d_in[0];
  const float* pair  = (const float*)d_in[1];
  const float* lmg   = (const float*)d_in[2];
  const float* lmb   = (const float*)d_in[3];
  const float* Wv    = (const float*)d_in[4];
  const float* bv    = (const float*)d_in[5];
  const float* lzg   = (const float*)d_in[6];
  const float* lzb   = (const float*)d_in[7];
  const float* Wb    = (const float*)d_in[8];
  const float* bb    = (const float*)d_in[9];
  const float* Wg    = (const float*)d_in[10];
  const float* bg    = (const float*)d_in[11];
  const float* Wp    = (const float*)d_in[12];
  float* out = (float*)d_out;

  char* ws = (char*)d_ws;
  const size_t SZ_MNHC = (size_t)SDIM * NDIM * HC * sizeof(bf16);   // 64 MB
  size_t off = 0;
  bf16* m_bf  = (bf16*)(ws + off); off += SZ_MNHC;                  // m (reused as o)
  bf16* vP    = (bf16*)(ws + off); off += SZ_MNHC;                  // packed B frags
  bf16* g_bf  = (bf16*)(ws + off); off += SZ_MNHC;
  float* b_f  = (float*)(ws + off); off += (size_t)NDIM * NDIM * HD * sizeof(float);
  bf16* w_bf  = (bf16*)(ws + off); off += (size_t)HD * NDIM * NDIM * sizeof(bf16);
  bf16* WvP   = (bf16*)(ws + off); off += (size_t)CMD * HC * sizeof(bf16);
  bf16* WgP   = (bf16*)(ws + off); off += (size_t)CMD * HC * sizeof(bf16);
  bf16* WpP   = (bf16*)(ws + off); off += (size_t)HC * CMD * sizeof(bf16);
  bf16* o_bf  = m_bf;  // m is dead after k_gemm_vg; reuse for gated output

  const int TB = 256;  // 8 wave32s per block

  // 1) msa LayerNorm -> bf16      (131072 rows, one wave per row)
  k_msa_ln<<<(SDIM * NDIM) / 8, TB, 0, stream>>>(msa, lmg, lmb, m_bf);

  // 2) pack weights into WMMA B-fragment layout (128 tiles each -> 16 blocks)
  k_pack_w<<<16, TB, 0, stream>>>(Wv, WvP, CMD, HC);
  k_pack_w<<<16, TB, 0, stream>>>(Wg, WgP, CMD, HC);
  k_pack_w<<<16, TB, 0, stream>>>(Wp, WpP, HC, CMD);

  // 3) v = m@Wv+bv (packed), g = sigmoid(m@Wg+bg)   (65536 waves)
  k_gemm_vg<<<(SDIM * NDIM / 32) * (HC / 16) / 8, TB, 0, stream>>>(
      m_bf, WvP, WgP, bv, bg, vP, g_bf);

  // 4) b = LN(pair)@Wb+bb      (262144 waves)
  k_pair_ln_b<<<(NDIM * NDIM) / 8, TB, 0, stream>>>(pair, lzg, lzb, Wb, bb, b_f);

  // 5) w = softmax_j(b), bf16 in [h][i][j]   (4096 waves)
  k_softmax<<<(NDIM * HD) / 8, TB, 0, stream>>>(b_f, w_bf);

  // 6) o = g * (w @ v) per (s,h)   (32768 waves, 32x32 macro-tiles)
  k_avg_gate<<<(SDIM * HD * (NDIM / 32)) / 8, TB, 0, stream>>>(w_bf, vP, g_bf, o_bf);

  // 7) out = o @ Wp  (f32)
  k_gemm_out<<<(SDIM * NDIM / 32) * (CMD / 16) / 8, TB, 0, stream>>>(o_bf, WpP, out);
}
